// ModulatedDeformConv2dResMultiBlock_70884140253364
// MI455X (gfx1250) — matrix operand order repaired and checked by reading, not compiled, not run on hardware
//
#include <hip/hip_runtime.h>

// ---------------------------------------------------------------------------
// CDNA5 (gfx1250) fused modulated-deformable-conv pipeline.
// All contractions run through v_wmma_f32_16x16x32_bf16 (fp32 accumulate).
// im2col (with bilinear deformable sampling) is staged in LDS as bf16 in the
// exact WMMA B-fragment order; weights are pre-packed in A-fragment order.
// Each wave owns M2 M-tiles so every pair of B fragments feeds 2*M2 WMMAs
// (2:1 b128-load:wmma ratio in the steady-state inner loop).
// ---------------------------------------------------------------------------

typedef __attribute__((ext_vector_type(16))) __bf16       v16bf;
typedef __attribute__((ext_vector_type(8)))  float        v8f;
typedef __attribute__((ext_vector_type(4)))  unsigned int v4u;

union Frag32B { v4u u[2]; v16bf bf; };

__device__ __forceinline__ unsigned short f2bf(float f) {
  unsigned u = __float_as_uint(f);
  u += 0x7FFFu + ((u >> 16) & 1u);          // round-to-nearest-even
  return (unsigned short)(u >> 16);
}

// Pack fp32 weight matrix (O_real rows, padded to M_total, K_total cols) into
// bf16 A-fragment order: [kstep][mtile][lane][j], where for 16-bit A fragments
//   row = 16*mtile + (lane&15)
//   K   = 32*kstep + 16*(j>>3) + 8*(lane>>4) + (j&7)     (ISA 05_wmma A table)
__global__ void dcn_pack_weights_kernel(const float* __restrict__ w,
                                        unsigned short* __restrict__ dst,
                                        int O_real, int M_total, int K_total) {
  const int idx = blockIdx.x * blockDim.x + threadIdx.x;
  const int total = M_total * K_total;
  if (idx >= total) return;
  const int Mt   = M_total >> 4;
  const int j    = idx & 15;
  const int lane = (idx >> 4) & 31;
  const int g    = idx >> 9;          // = s*Mt + mt
  const int mt   = g % Mt;
  const int s    = g / Mt;
  const int row  = mt * 16 + (lane & 15);
  const int K    = s * 32 + ((j >> 3) << 4) + ((lane >> 4) << 3) + (j & 7);
  const float v  = (row < O_real) ? w[(size_t)row * K_total + K] : 0.0f;
  dst[idx] = f2bf(v);
}

// Conv-as-GEMM: out[o, p] = sum_{c,k2} w[o, c*K2+k2] * cols[c,k2,p]
//   DEFORM=false : regular 'same' conv taps (offset convs, 1x1 sub-conv)
//   DEFORM=true  : bilinear deformable sampling modulated by sigmoid mask,
//                  epilogue additionally adds xs and applies ReLU.
// Tile: 32 pixels x (16*M_TILES) channels per block; M_TILES/M2 waves, each
// wave owning M2 16-row M-tiles. K chunked by CCHUNK input channels so LDS
// stays <= 50KB. O_REAL templated so full-M instances get a branch-free
// epilogue (guard elided at compile time).
template <int KSZ, int CCHUNK, int M_TILES, int M2, bool DEFORM, int O_REAL>
__global__ __launch_bounds__((M_TILES / M2) * 32)
void dcn_gemm_kernel(const float* __restrict__ x,
                     const float* __restrict__ om,
                     const unsigned short* __restrict__ wpk,
                     const float* __restrict__ bias,
                     const float* __restrict__ xs,
                     float* __restrict__ out,
                     long out_b_stride, int ch_off) {
  constexpr int K2      = KSZ * KSZ;
  constexpr int PAD     = KSZ / 2;
  constexpr int CIN     = 64;
  constexpr int H = 56, W = 56, NP = H * W;
  constexpr int NCH     = CIN / CCHUNK;        // channel chunks
  constexpr int KCH     = CCHUNK * K2;         // K per chunk (multiple of 32)
  constexpr int KSTEPS  = KCH / 32;
  constexpr int NT_IMG  = NP / 32;             // 98 pixel tiles per image
  constexpr int NWAVES  = M_TILES / M2;
  constexpr int NTHREADS = NWAVES * 32;
  constexpr bool FULL_M  = (O_REAL == M_TILES * 16);

  extern __shared__ unsigned short colsT[];    // [32 pixels][KCH] bf16 bits

  const int tid  = threadIdx.x;
  const int lane = tid & 31;
  const int wv   = tid >> 5;                   // wave id
  const int b    = blockIdx.x / NT_IMG;
  const int pt   = blockIdx.x % NT_IMG;
  const int p0   = pt * 32;

  v8f acc[M2][2];
#pragma unroll
  for (int m = 0; m < M2; ++m)
#pragma unroll
    for (int t = 0; t < 2; ++t)
      acc[m][t] = v8f{0.f, 0.f, 0.f, 0.f, 0.f, 0.f, 0.f, 0.f};

  const float* xb = x + (size_t)b * CIN * NP;

  for (int ch = 0; ch < NCH; ++ch) {
    if (ch) __syncthreads();                   // prior chunk fully consumed
    const int c0 = ch * CCHUNK;

    // ---- build bf16 cols tile in LDS -------------------------------------
    for (int pair = tid; pair < K2 * 32; pair += NTHREADS) {
      const int k2 = pair >> 5;
      const int pn = pair & 31;
      const int pg = p0 + pn;
      const int oy = pg / W;
      const int ox = pg - oy * W;
      const int ky = k2 / KSZ;
      const int kx = k2 - ky * KSZ;
      unsigned short* dst = &colsT[pn * KCH + k2];
      if constexpr (DEFORM) {
        const size_t ob = (size_t)b * 3 * K2 * NP + pg;
        const float dy  = om[ob + (size_t)k2 * NP];
        const float dx  = om[ob + (size_t)(K2 + k2) * NP];
        const float ml  = om[ob + (size_t)(2 * K2 + k2) * NP];
        const float msk = 1.0f / (1.0f + __expf(-ml));
        const float fy  = (float)(oy + ky - PAD) + dy;
        const float fx  = (float)(ox + kx - PAD) + dx;
        const float y0f = floorf(fy), x0f = floorf(fx);
        const float ty = fy - y0f, tx = fx - x0f;
        const int y0 = (int)y0f, x0 = (int)x0f;
        const int y1 = y0 + 1,  x1 = x0 + 1;
        const bool vy0 = (y0 >= 0) & (y0 < H), vy1 = (y1 >= 0) & (y1 < H);
        const bool vx0 = (x0 >= 0) & (x0 < W), vx1 = (x1 >= 0) & (x1 < W);
        const float w00 = (vy0 & vx0) ? (1.f - ty) * (1.f - tx) * msk : 0.f;
        const float w01 = (vy0 & vx1) ? (1.f - ty) * tx * msk : 0.f;
        const float w10 = (vy1 & vx0) ? ty * (1.f - tx) * msk : 0.f;
        const float w11 = (vy1 & vx1) ? ty * tx * msk : 0.f;
        const int y0c = min(max(y0, 0), H - 1), y1c = min(max(y1, 0), H - 1);
        const int x0c = min(max(x0, 0), W - 1), x1c = min(max(x1, 0), W - 1);
        const int i00 = y0c * W + x0c, i01 = y0c * W + x1c;
        const int i10 = y1c * W + x0c, i11 = y1c * W + x1c;
        const float* pc = xb + (size_t)c0 * NP;
#pragma unroll 4
        for (int c = 0; c < CCHUNK; ++c) {
          const float v = w00 * pc[i00] + w01 * pc[i01] +
                          w10 * pc[i10] + w11 * pc[i11];
          dst[c * K2] = f2bf(v);
          pc += NP;
        }
      } else {
        const int yy = oy + ky - PAD, xx = ox + kx - PAD;
        const bool valid = (yy >= 0) & (yy < H) & (xx >= 0) & (xx < W);
        const int ii = valid ? yy * W + xx : 0;
        const float sc = valid ? 1.f : 0.f;
        const float* pc = xb + (size_t)c0 * NP;
#pragma unroll 4
        for (int c = 0; c < CCHUNK; ++c) {
          dst[c * K2] = f2bf(sc * pc[ii]);
          pc += NP;
        }
      }
    }
    __syncthreads();

    // ---- WMMA GEMM over this K chunk -------------------------------------
    const unsigned short* wbase = wpk + (size_t)ch * KSTEPS * M_TILES * 512;
#pragma unroll 1
    for (int s = 0; s < KSTEPS; ++s) {
      // A fragments: coalesced 16B loads per lane from packed weights
      Frag32B a[M2];
#pragma unroll
      for (int m = 0; m < M2; ++m) {
        const v4u* ap = (const v4u*)(wbase +
            ((size_t)s * M_TILES + (wv * M2 + m)) * 512 + (size_t)lane * 16);
        a[m].u[0] = ap[0];
        a[m].u[1] = ap[1];
      }
      // B fragments: lanes 0-15 take K s*32+{0..7,16..23}, lanes 16-31 take
      // s*32+{8..15,24..31}; column N = lane&15 -> 2x ds_load_b128 each.
      const int kb   = s * 32 + ((lane >> 4) << 3);
      const int nrow = lane & 15;
      Frag32B fb0, fb1;
      const v4u* bp0 = (const v4u*)&colsT[(size_t)nrow * KCH + kb];
      const v4u* bp1 = (const v4u*)&colsT[(size_t)(nrow + 16) * KCH + kb];
      fb0.u[0] = bp0[0]; fb0.u[1] = bp0[2];
      fb1.u[0] = bp1[0]; fb1.u[1] = bp1[2];
#pragma unroll
      for (int m = 0; m < M2; ++m) {
        acc[m][0] = __builtin_amdgcn_wmma_f32_16x16x32_bf16(
            false, a[m].bf, false, fb0.bf, (short)0, acc[m][0], false, false);
        acc[m][1] = __builtin_amdgcn_wmma_f32_16x16x32_bf16(
            false, a[m].bf, false, fb1.bf, (short)0, acc[m][1], false, false);
      }
    }
  }

  // ---- epilogue: D layout -> bias (+ xs + relu if DEFORM) + store ---------
  const int half = lane >> 4;                  // VGPR v: M = v + 8*half
  const int ncol = lane & 15;                  // N = lane & 15
#pragma unroll
  for (int m = 0; m < M2; ++m) {
#pragma unroll
    for (int t = 0; t < 2; ++t) {
      const int p = p0 + t * 16 + ncol;
#pragma unroll
      for (int v = 0; v < 8; ++v) {
        const int o = (wv * M2 + m) * 16 + v + 8 * half;
        if (FULL_M || o < O_REAL) {            // guard elided when padding==0
          float val = acc[m][t][v] + bias[o];
          if constexpr (DEFORM) {
            val += xs[((size_t)b * 128 + o) * NP + p];
            val = fmaxf(val, 0.0f);
          }
          out[(size_t)b * out_b_stride + (size_t)(ch_off + o) * NP + p] = val;
        }
      }
    }
  }
}

// ---------------------------------------------------------------------------
extern "C" void kernel_launch(void* const* d_in, const int* in_sizes, int n_in,
                              void* d_out, int out_size, void* d_ws,
                              size_t ws_size, hipStream_t stream) {
  (void)in_sizes; (void)n_in; (void)out_size; (void)ws_size;
  const float* x      = (const float*)d_in[0];
  const float* w3     = (const float*)d_in[1];
  const float* b3     = (const float*)d_in[2];
  const float* off3_w = (const float*)d_in[3];
  const float* off3_b = (const float*)d_in[4];
  const float* w5     = (const float*)d_in[5];
  const float* b5     = (const float*)d_in[6];
  const float* off5_w = (const float*)d_in[7];
  const float* off5_b = (const float*)d_in[8];
  const float* sub_w  = (const float*)d_in[9];
  const float* sub_b  = (const float*)d_in[10];
  float* out = (float*)d_out;

  constexpr int B = 8, NP = 56 * 56;

  // workspace carve-out (~24 MB), 256B aligned
  char* ws = (char*)d_ws;
  size_t off = 0;
  auto carve = [&](size_t bytes) -> char* {
    char* p = ws + off;
    off = (off + bytes + 255) & ~(size_t)255;
    return p;
  };
  float* om3 = (float*)carve((size_t)B * 27 * NP * 4);   // (B,27,H,W)
  float* om5 = (float*)carve((size_t)B * 75 * NP * 4);   // (B,75,H,W)
  float* xs  = (float*)carve((size_t)B * 128 * NP * 4);  // 1x1 conv result
  unsigned short* w3p = (unsigned short*)carve((size_t)128 * 576 * 2);
  unsigned short* w5p = (unsigned short*)carve((size_t)128 * 1600 * 2);
  unsigned short* o3p = (unsigned short*)carve((size_t)32 * 576 * 2);
  unsigned short* o5p = (unsigned short*)carve((size_t)80 * 1600 * 2);
  unsigned short* sbp = (unsigned short*)carve((size_t)128 * 64 * 2);

  auto pack = [&](const float* w, unsigned short* dst, int Oreal, int Mtot,
                  int Ktot) {
    const int total = Mtot * Ktot;
    dcn_pack_weights_kernel<<<(total + 255) / 256, 256, 0, stream>>>(
        w, dst, Oreal, Mtot, Ktot);
  };
  pack(w3,     w3p, 128, 128, 576);
  pack(w5,     w5p, 128, 128, 1600);
  pack(off3_w, o3p,  27,  32, 576);
  pack(off5_w, o5p,  75,  80, 1600);
  pack(sub_w,  sbp, 128, 128, 64);

  const int grid = B * (NP / 32);  // 784 blocks, 32 pixels each

  // stage 1: offset convs + 1x1 sub-conv (regular conv as WMMA GEMM)
  dcn_gemm_kernel<3, 64, 2, 1, false, 27><<<grid, 64, 32 * 576 * 2, stream>>>(
      x, nullptr, o3p, off3_b, nullptr, om3, (long)27 * NP, 0);
  dcn_gemm_kernel<5, 32, 5, 1, false, 75><<<grid, 160, 32 * 800 * 2, stream>>>(
      x, nullptr, o5p, off5_b, nullptr, om5, (long)75 * NP, 0);
  dcn_gemm_kernel<1, 64, 8, 2, false, 128><<<grid, 128, 32 * 64 * 2, stream>>>(
      x, nullptr, sbp, sub_b, nullptr, xs, (long)128 * NP, 0);

  // stage 2: modulated deformable convs, fused bias + xs + relu + concat
  dcn_gemm_kernel<3, 64, 8, 2, true, 128><<<grid, 128, 32 * 576 * 2, stream>>>(
      x, om3, w3p, b3, xs, out, (long)256 * NP, 0);
  dcn_gemm_kernel<5, 32, 8, 2, true, 128><<<grid, 128, 32 * 800 * 2, stream>>>(
      x, om5, w5p, b5, xs, out, (long)256 * NP, 128);
}
